// PairwiseStructureBuilder_78855599555041
// MI455X (gfx1250) — 8-bit matrix operand order repaired and checked by reading, compiled, not run on hardware
//
#include <hip/hip_runtime.h>
#include <math.h>


#define B_ 64
#define N_ 256
#define C_ 64

typedef int v8i __attribute__((ext_vector_type(8)));

static constexpr size_t BNN = (size_t)B_ * N_ * N_;
static constexpr size_t BN  = (size_t)B_ * N_;

static constexpr size_t O_PAIR  = 0;
static constexpr size_t O_DX    = 1 * BNN;
static constexpr size_t O_DY    = 2 * BNN;
static constexpr size_t O_DIST  = 3 * BNN;
static constexpr size_t O_NDIST = 4 * BNN;
static constexpr size_t O_CONT  = 5 * BNN;
static constexpr size_t O_DISTB = 11 * BNN;
static constexpr size_t O_DIRB  = 12 * BNN;
static constexpr size_t O_ROLE  = 13 * BNN;
static constexpr size_t O_RPAIR = 13 * BNN + BN;
static constexpr size_t O_HOPD  = 14 * BNN + BN;
static constexpr size_t O_ET    = 15 * BNN + BN;
static constexpr size_t O_SPB   = 16 * BNN + BN;
static constexpr size_t O_DEG   = 17 * BNN + BN;
static constexpr size_t O_CT    = 17 * BNN + 2 * BN;
static constexpr size_t O_SCT   = 17 * BNN + 3 * BN;

__constant__ float c_dist_edges[6] = {0.5f, 0.9f, 1.25f, 1.75f, 2.5f, 3.5f};

__global__ void node_kernel(const float* __restrict__ coords,
                            const int* __restrict__ hop,
                            const int* __restrict__ valid,
                            const int* __restrict__ adj,
                            const float* __restrict__ cell,
                            float* __restrict__ out,
                            float* __restrict__ scaleWs,
                            int* __restrict__ roleWs,
                            int* __restrict__ ctIdWs,
                            unsigned char* __restrict__ ctValWs) {
    const int b = blockIdx.x;
    const int i = threadIdx.x;
    __shared__ float ssum[256];
    __shared__ int   scnt[256];

    const int   v  = (valid[b * N_ + i] != 0);
    const float x  = coords[((size_t)b * N_ + i) * 2 + 0];
    const float y  = coords[((size_t)b * N_ + i) * 2 + 1];

    const int   nb = v && (i != 0);
    ssum[i] = nb ? sqrtf(x * x + y * y + 1e-8f) : 0.0f;
    scnt[i] = nb;
    __syncthreads();
    for (int s = 128; s > 0; s >>= 1) {
        if (i < s) { ssum[i] += ssum[i + s]; scnt[i] += scnt[i + s]; }
        __syncthreads();
    }
    if (i == 0) {
        float cnt = (float)(scnt[0] > 1 ? scnt[0] : 1);
        float sc  = ssum[0] / cnt;
        sc = (sc > 0.0f) ? sc : 1.0f;
        sc = fmaxf(sc, 1e-6f);
        scaleWs[b] = sc;
    }

    const int* __restrict__ row = adj + ((size_t)b * N_ + i) * N_;
    int deg = 0;
    for (int j = 0; j < N_; ++j) deg += (row[j] != 0);
    deg -= v;
    deg = min(max(deg, 0), 16);
    const int degId = v ? deg + 1 : 0;

    const float* __restrict__ ci = cell + ((size_t)b * N_ + i) * C_;
    float s = 0.0f, mx = -INFINITY;
    int am = 0;
    for (int c = 0; c < C_; ++c) {
        float vv = ci[c];
        s += fabsf(vv);
        if (vv > mx) { mx = vv; am = c; }
    }
    const int ctValid = (s > 0.0f) && v;
    const int ctId    = ctValid ? (am + 1) : 0;

    const int role = min(max(hop[b * N_ + i], 0), 2);

    out[O_ROLE + (size_t)b * N_ + i] = (float)role;
    out[O_DEG  + (size_t)b * N_ + i] = (float)degId;
    out[O_CT   + (size_t)b * N_ + i] = (float)ctId;
    roleWs[b * N_ + i]  = role;
    ctIdWs[b * N_ + i]  = ctId;
    ctValWs[b * N_ + i] = (unsigned char)ctValid;
}

__global__ void pack_kernel(const int* __restrict__ adj,
                            unsigned char* __restrict__ A8,
                            unsigned char* __restrict__ A8T) {
    const size_t idx = (size_t)blockIdx.x * 256 + threadIdx.x;
    const int b   = (int)(idx / ((size_t)N_ * N_));
    const int pos = (int)(idx % ((size_t)N_ * N_));
    const int i   = pos / N_;
    const int j   = pos % N_;
    const unsigned char v = (unsigned char)((adj[idx] != 0) || (i == j));
    A8[idx] = v;
    A8T[(size_t)b * N_ * N_ + (size_t)j * N_ + i] = v;
}

__global__ void bmm_kernel(const unsigned char* __restrict__ S,
                           const unsigned char* __restrict__ T,
                           unsigned char* __restrict__ D) {
    __shared__ __align__(16) unsigned char sA[16 * N_];

    const int lane = threadIdx.x;
    const int wave = threadIdx.y;
    const int b    = blockIdx.y;
    const int tile = blockIdx.x * 8 + wave;
    const int tm   = tile >> 4;
    const int tn   = tile & 15;

    const unsigned char* __restrict__ Sb = S + (size_t)b * N_ * N_;
    const unsigned char* __restrict__ Tb = T + (size_t)b * N_ * N_;

    {
        const int t = wave * 32 + lane;
        const unsigned char* g = Sb + (size_t)tm * 16 * N_ + (size_t)t * 16;
        unsigned ldsOff = (unsigned)(size_t)(&sA[t * 16]);
        asm volatile("global_load_async_to_lds_b128 %0, %1, off"
                     :: "v"(ldsOff), "v"(g)
                     : "memory");
        asm volatile("s_wait_asynccnt 0" ::: "memory");
    }
    __syncthreads();

    const int m    = lane & 15;
    const int hi   = (lane >> 4) & 1;
    const int colB = tn * 16 + m;

    v8i acc = {0, 0, 0, 0, 0, 0, 0, 0};

    for (int kt = 0; kt < 4; ++kt) {
        const int kb = kt * 64;
        if (kt < 3)
            __builtin_prefetch(Tb + (size_t)colB * N_ + kb + 64, 0, 0);
        v8i a, bf;
#pragma unroll
        for (int vv = 0; vv < 8; ++vv) {
            const int ka = kb + 16 * (vv >> 1) + 4 * (vv & 1) + hi * 8;
            a[vv] = *(const int*)(sA + m * N_ + ka);
            const int kbb = kb + 16 * (vv >> 1) + 4 * (vv & 1) + hi * 8;
            bf[vv] = *(const int*)(Tb + (size_t)colB * N_ + kbb);
        }
        acc = __builtin_amdgcn_wmma_i32_16x16x64_iu8(
             false, a,  false, bf, acc,
             false,  false);
    }

#pragma unroll
    for (int vv = 0; vv < 8; ++vv) {
        const int r = tm * 16 + vv + hi * 8;
        const int c = tn * 16 + m;
        D[(size_t)b * N_ * N_ + (size_t)r * N_ + c] =
            (unsigned char)(acc[vv] > 0 ? 1 : 0);
    }
}

__global__ void pair_kernel(const float* __restrict__ coords,
                            const int* __restrict__ valid,
                            const int* __restrict__ adj,
                            const float* __restrict__ scaleWs,
                            const int* __restrict__ roleWs,
                            const int* __restrict__ ctIdWs,
                            const unsigned char* __restrict__ ctValWs,
                            const unsigned char* __restrict__ R2,
                            const unsigned char* __restrict__ R3,
                            const unsigned char* __restrict__ R4,
                            float* __restrict__ out) {
    const int i = blockIdx.x;
    const int b = blockIdx.y;
    const int j = threadIdx.x;
    const size_t bi  = (size_t)b * N_ + i;
    const size_t bj  = (size_t)b * N_ + j;
    const size_t bij = bi * N_ + j;

    const float xi = coords[bi * 2 + 0], yi = coords[bi * 2 + 1];
    const float xj = coords[bj * 2 + 0], yj = coords[bj * 2 + 1];
    const int vi = (valid[bi] != 0), vj = (valid[bj] != 0);
    const bool validPair = vi && vj;
    const bool adjv = (adj[bij] != 0);
    const bool eye  = (i == j);
    const bool pm   = adjv && validPair;
    const float pmf = pm ? 1.0f : 0.0f;

    const float dx = xj - xi;
    const float dy = yj - yi;
    float dist = sqrtf(dx * dx + dy * dy + 1e-8f);
    if (eye) dist = 0.0f;

    const float scale = scaleWs[b];
    const float ndx = dx / scale, ndy = dy / scale, nd = dist / scale;
    const float lg = log1pf(nd);

    int db = 0;
#pragma unroll
    for (int e = 0; e < 6; ++e) db += (c_dist_edges[e] < nd);
    db += 1;
    if (eye) db = 0;
    if (!validPair) db = 8;

    const float TWO_PI = 6.28318530717958647692f;
    float ang = atan2f(dy, dx) + TWO_PI;
    ang = ang - floorf(ang / TWO_PI) * TWO_PI;
    int dirb = (int)floorf(ang / (TWO_PI / 8.0f)) + 1;
    dirb = min(dirb, 8);
    if (eye) dirb = 0;
    if (!validPair) dirb = 9;

    const int ri = roleWs[bi], rj = roleWs[bj];
    const int rp = validPair ? (ri * 3 + rj) : 9;
    const int hd = validPair ? abs(rj - ri) : 3;

    const bool center = pm && !eye && ((ri == 0) || (rj == 0));
    const bool same   = pm && !eye && !center && (ri == rj);
    const bool cross  = pm && !eye && !center && (ri != rj);
    int et = 4;
    if (eye && validPair) et = 0;
    if (center) et = 1;
    if (same)   et = 2;
    if (cross)  et = 3;

    int spb;
    if (eye)                spb = 0;
    else if (adjv)          spb = 1;
    else if (R2[bij])       spb = 2;
    else if (R3[bij])       spb = 3;
    else if (R4[bij])       spb = 4;
    else                    spb = 5;
    if (!validPair)         spb = 5;

    const bool known = ctValWs[bi] && ctValWs[bj];
    int sct = 0;
    if (known) sct = (ctIdWs[bi] == ctIdWs[bj]) ? 1 : 2;

    out[O_PAIR  + bij] = pmf;
    out[O_DX    + bij] = dx;
    out[O_DY    + bij] = dy;
    out[O_DIST  + bij] = dist;
    out[O_NDIST + bij] = nd;
    const size_t oc = O_CONT + bij * 6;
    out[oc + 0] = ndx * pmf;
    out[oc + 1] = ndy * pmf;
    out[oc + 2] = nd * pmf;
    out[oc + 3] = lg * pmf;
    out[oc + 4] = fabsf(ndx) * pmf;
    out[oc + 5] = fabsf(ndy) * pmf;
    out[O_DISTB + bij] = (float)db;
    out[O_DIRB  + bij] = (float)dirb;
    out[O_RPAIR + bij] = (float)rp;
    out[O_HOPD  + bij] = (float)hd;
    out[O_ET    + bij] = (float)et;
    out[O_SPB   + bij] = (float)spb;
    out[O_SCT   + bij] = (float)sct;
}

extern "C" void kernel_launch(void* const* d_in, const int* in_sizes, int n_in,
                              void* d_out, int out_size, void* d_ws, size_t ws_size,
                              hipStream_t stream) {
    const float* coords = (const float*)d_in[0];
    const int*   hop    = (const int*)d_in[1];
    const int*   valid  = (const int*)d_in[2];
    const int*   adj    = (const int*)d_in[3];
    const float* cell   = (const float*)d_in[4];
    float* out = (float*)d_out;

    char* w = (char*)d_ws;
    const size_t MAT = BNN;
    unsigned char* A8   = (unsigned char*)(w + 0 * MAT);
    unsigned char* A8T  = (unsigned char*)(w + 1 * MAT);
    unsigned char* R2   = (unsigned char*)(w + 2 * MAT);
    unsigned char* R3   = (unsigned char*)(w + 3 * MAT);
    unsigned char* R4   = (unsigned char*)(w + 4 * MAT);
    float*         scl  = (float*)(w + 5 * MAT);
    int*           role = (int*)(w + 5 * MAT + 1024);
    int*           ctId = (int*)(w + 5 * MAT + 1024 + BN * 4);
    unsigned char* ctVl = (unsigned char*)(w + 5 * MAT + 1024 + BN * 8);

    node_kernel<<<B_, N_, 0, stream>>>(coords, hop, valid, adj, cell,
                                       out, scl, role, ctId, ctVl);
    pack_kernel<<<(unsigned)(BNN / 256), 256, 0, stream>>>(adj, A8, A8T);

    dim3 bmmGrid(32, B_);
    dim3 bmmBlock(32, 8);
    bmm_kernel<<<bmmGrid, bmmBlock, 0, stream>>>(A8, A8T, R2);
    bmm_kernel<<<bmmGrid, bmmBlock, 0, stream>>>(R2, A8T, R3);
    bmm_kernel<<<bmmGrid, bmmBlock, 0, stream>>>(R3, A8T, R4);

    dim3 pairGrid(N_, B_);
    pair_kernel<<<pairGrid, N_, 0, stream>>>(coords, valid, adj, scl, role,
                                             ctId, ctVl, R2, R3, R4, out);
}
